// NNUENetworkDense_24026047053857
// MI455X (gfx1250) — matrix operand order repaired and checked by reading, not compiled, not run on hardware
//
#include <hip/hip_runtime.h>
#include <hip/hip_bf16.h>

// ---------------------------------------------------------------------------
// NNUE forward pass on gfx1250 (MI455X).
//   wh = clip(white_onehot @ ft_w^T + ft_b, 0, 1)   [2048 x 256]
//   bh = clip(black_onehot @ ft_w^T + ft_b, 0, 1)   [2048 x 256]
//   hidden = stm ? [wh,bh] : [bh,wh]                [2048 x 512]
//   x = clip(hidden @ l1_w^T + b); x = clip(x @ l2_w^T + b); out = x @ l3_w^T + b
//
// Roofline: one-hot reads = 671 MB -> ~29 us at 23.3 TB/s (HBM floor).
// M=32 rows/workgroup halves L2 re-streaming of f16 ft_w (one B fragment
// feeds two WMMAs). Fragment loads are software-pipelined one K-step ahead
// so WMMAs never wait on loads issued in the same step.
// ---------------------------------------------------------------------------

typedef __attribute__((ext_vector_type(16))) _Float16 v16h;
typedef __attribute__((ext_vector_type(8)))  _Float16 half8;
typedef __attribute__((ext_vector_type(8)))  float    v8f;

#define NN_B      2048
#define NN_IN     40960
#define NN_H      256
#define NN_H2     512

#define MROWS     32             // batch rows per workgroup (2 WMMA M-tiles)
#define KC        256            // K-chunk staged per barrier
#define NWMMA     (KC / 32)      // 8 K-steps per staged chunk
#define NCHUNK    (NN_IN / KC)   // 160 chunks

// ---------------------------------------------------------------------------
// Kernel 0: f32 -> f16 conversion of ft_w (256 x 40960), 8 elems / thread.
// ---------------------------------------------------------------------------
__global__ __launch_bounds__(256)
void nnue_cvt_f16_kernel(const float* __restrict__ src,
                         _Float16* __restrict__ dst, int n8) {
    int i = blockIdx.x * blockDim.x + threadIdx.x;
    if (i >= n8) return;
    const float4 a = *(const float4*)(src + (size_t)i * 8);
    const float4 b = *(const float4*)(src + (size_t)i * 8 + 4);
    half8 h;
    h[0] = (_Float16)a.x; h[1] = (_Float16)a.y;
    h[2] = (_Float16)a.z; h[3] = (_Float16)a.w;
    h[4] = (_Float16)b.x; h[5] = (_Float16)b.y;
    h[6] = (_Float16)b.z; h[7] = (_Float16)b.w;
    *(half8*)(dst + (size_t)i * 8) = h;
}

// ---------------------------------------------------------------------------
// Kernel 1: feature transformer GEMM via WMMA.
//   grid  = (64 batch strips of 32 rows, 2 sides), block = 512 = 16 waves.
//   Wave w owns hidden columns w*16..w*16+15 for BOTH 16-row M-sub-tiles.
//   Double-buffered LDS staging (one barrier per K-chunk); A/B fragments
//   pipelined one K-step ahead in registers.
// ---------------------------------------------------------------------------
__global__ __launch_bounds__(512)
void nnue_ft_wmma_kernel(const float* __restrict__ white,
                         const float* __restrict__ black,
                         const float* __restrict__ stm,
                         const _Float16* __restrict__ ftw_h,  // [256][40960] f16
                         const float* __restrict__ ftb,       // [256]
                         _Float16* __restrict__ hidden)       // [2048][512] f16
{
    const int strip = blockIdx.x;      // batch rows strip*32 .. +31
    const int side  = blockIdx.y;      // 0 = white, 1 = black
    const float* __restrict__ A =
        (side == 0 ? white : black) + (size_t)strip * MROWS * NN_IN;

    const int tid  = threadIdx.x;
    const int wave = tid >> 5;         // 0..15 -> hidden-column tile
    const int lane = tid & 31;
    const int lh   = lane >> 4;        // lane half (0 or 1)
    const int nl   = lane & 15;

    __shared__ _Float16 As[2][MROWS][KC];  // double-buffered, 32 KB total

    v8f c0 = {};                       // rows  0..15 of the strip
    v8f c1 = {};                       // rows 16..31 of the strip

    // B fragment source: row (wave*16 + nl) of f16 ft_w, K-major contiguous.
    const _Float16* __restrict__ brow = ftw_h + (size_t)(wave * 16 + nl) * NN_IN;

    // staging map: 512 threads x 16 f32 <-> 32 x KC tile
    const int sr = tid >> 4;           // staging row   0..31
    const int sc = (tid & 15) * 16;    // staging col   0,16,..,240
    const float* __restrict__ arow = A + (size_t)sr * NN_IN + sc;

    // prologue: load chunk 0 into registers (4 x b128)
    float4 r0 = *(const float4*)(arow);
    float4 r1 = *(const float4*)(arow + 4);
    float4 r2 = *(const float4*)(arow + 8);
    float4 r3 = *(const float4*)(arow + 12);

    union Frag { v16h v; half8 h[2]; };

    for (int ch = 0; ch < NCHUNK; ++ch) {
        // commit staged registers (f32 -> f16) into this chunk's buffer
        half8 h0, h1;
        h0[0] = (_Float16)r0.x; h0[1] = (_Float16)r0.y;
        h0[2] = (_Float16)r0.z; h0[3] = (_Float16)r0.w;
        h0[4] = (_Float16)r1.x; h0[5] = (_Float16)r1.y;
        h0[6] = (_Float16)r1.z; h0[7] = (_Float16)r1.w;
        h1[0] = (_Float16)r2.x; h1[1] = (_Float16)r2.y;
        h1[2] = (_Float16)r2.z; h1[3] = (_Float16)r2.w;
        h1[4] = (_Float16)r3.x; h1[5] = (_Float16)r3.y;
        h1[6] = (_Float16)r3.z; h1[7] = (_Float16)r3.w;
        *(half8*)(&As[ch & 1][sr][sc])     = h0;
        *(half8*)(&As[ch & 1][sr][sc + 8]) = h1;
        __syncthreads();               // LDS release + acquire (s_wait_dscnt)

        // issue next chunk's staging loads now; they retire under the WMMAs
        if (ch + 1 < NCHUNK) {
            const float* __restrict__ nxt = arow + (size_t)(ch + 1) * KC;
            r0 = *(const float4*)(nxt);
            r1 = *(const float4*)(nxt + 4);
            r2 = *(const float4*)(nxt + 8);
            r3 = *(const float4*)(nxt + 12);
            __builtin_prefetch(brow + (size_t)(ch + 1) * KC, 0, 1);
        }

        const _Float16* __restrict__ ab0 = &As[ch & 1][nl][0];       // M-tile 0
        const _Float16* __restrict__ ab1 = &As[ch & 1][16 + nl][0];  // M-tile 1
        const _Float16* __restrict__ bb  = brow + (size_t)ch * KC;

        // -------- fragment pipeline: preload K-step 0 --------------------
        Frag ub, ua0, ua1;
        ub.h[0]  = *(const half8*)(bb + 8 * lh);
        ub.h[1]  = *(const half8*)(bb + 16 + 8 * lh);
        ua0.h[0] = *(const half8*)(ab0 + 8 * lh);
        ua0.h[1] = *(const half8*)(ab0 + 16 + 8 * lh);
        ua1.h[0] = *(const half8*)(ab1 + 8 * lh);
        ua1.h[1] = *(const half8*)(ab1 + 16 + 8 * lh);

#pragma unroll
        for (int i = 0; i < NWMMA - 1; ++i) {
            // issue K-step i+1's loads before K-step i's WMMAs
            Frag nb, na0, na1;
            const int o = (i + 1) * 32;
            nb.h[0]  = *(const half8*)(bb + o + 8 * lh);
            nb.h[1]  = *(const half8*)(bb + o + 16 + 8 * lh);
            na0.h[0] = *(const half8*)(ab0 + o + 8 * lh);
            na0.h[1] = *(const half8*)(ab0 + o + 16 + 8 * lh);
            na1.h[0] = *(const half8*)(ab1 + o + 8 * lh);
            na1.h[1] = *(const half8*)(ab1 + o + 16 + 8 * lh);

            c0 = __builtin_amdgcn_wmma_f32_16x16x32_f16(
                     false, ua0.v, false, ub.v, (short)0, c0, false, false);
            c1 = __builtin_amdgcn_wmma_f32_16x16x32_f16(
                     false, ua1.v, false, ub.v, (short)0, c1, false, false);

            ub = nb; ua0 = na0; ua1 = na1;   // rotate pipeline registers
        }
        // drain: last K-step of the chunk
        c0 = __builtin_amdgcn_wmma_f32_16x16x32_f16(
                 false, ua0.v, false, ub.v, (short)0, c0, false, false);
        c1 = __builtin_amdgcn_wmma_f32_16x16x32_f16(
                 false, ua1.v, false, ub.v, (short)0, c1, false, false);
    }

    // Epilogue: bias + clamp + stm-ordered scatter into hidden (f16).
    // C layout: lane -> N = nl (fixed), VGPR r -> M = r + 8*lh.
    const int col  = wave * 16 + nl;
    const float bias = ftb[col];
#pragma unroll
    for (int mt = 0; mt < 2; ++mt) {
        const v8f c = mt ? c1 : c0;
#pragma unroll
        for (int r = 0; r < 8; ++r) {
            const int row = strip * MROWS + mt * 16 + r + 8 * lh;
            float v = c[r] + bias;
            v = fminf(fmaxf(v, 0.0f), 1.0f);
            const bool white_first = (stm[row] != 0.0f);
            const int base = ((side == 0) == white_first) ? 0 : NN_H;
            hidden[(size_t)row * NN_H2 + base + col] = (_Float16)v;
        }
    }
}

// ---------------------------------------------------------------------------
// Kernel 2: tiny tail layers. One wave (32 threads) per batch row.
// ---------------------------------------------------------------------------
__global__ __launch_bounds__(32)
void nnue_tail_kernel(const _Float16* __restrict__ hidden,  // [2048][512]
                      const float* __restrict__ l1w,        // [32][512]
                      const float* __restrict__ l1b,        // [32]
                      const float* __restrict__ l2w,        // [32][32]
                      const float* __restrict__ l2b,        // [32]
                      const float* __restrict__ l3w,        // [32]
                      const float* __restrict__ l3b,        // [1]
                      float* __restrict__ out)              // [2048]
{
    const int b = blockIdx.x;
    const int j = threadIdx.x;                 // 0..31, neuron index

    __shared__ _Float16 sh[NN_H2];
    __shared__ float s1[32];

    const _Float16* __restrict__ hrow = hidden + (size_t)b * NN_H2;
    for (int t = j; t < NN_H2 / 8; t += 32)    // 2 iters: 16-B vector copies
        *(half8*)(sh + t * 8) = *(const half8*)(hrow + t * 8);
    __syncthreads();

    // layer 1: 512 -> 32, clip
    float acc = l1b[j];
    const float* __restrict__ w1 = l1w + (size_t)j * NN_H2;
#pragma unroll 8
    for (int k = 0; k < NN_H2; ++k)
        acc += (float)sh[k] * w1[k];
    s1[j] = fminf(fmaxf(acc, 0.0f), 1.0f);
    __syncthreads();

    // layer 2: 32 -> 32, clip
    float acc2 = l2b[j];
    const float* __restrict__ w2 = l2w + (size_t)j * 32;
#pragma unroll
    for (int k = 0; k < 32; ++k)
        acc2 += s1[k] * w2[k];
    const float x2 = fminf(fmaxf(acc2, 0.0f), 1.0f);

    // layer 3: 32 -> 1, wave32 shuffle reduction
    float v = x2 * l3w[j];
    for (int off = 16; off > 0; off >>= 1)
        v += __shfl_down(v, off, 32);
    if (j == 0)
        out[b] = v + l3b[0];
}

// ---------------------------------------------------------------------------
extern "C" void kernel_launch(void* const* d_in, const int* in_sizes, int n_in,
                              void* d_out, int out_size, void* d_ws, size_t ws_size,
                              hipStream_t stream) {
    const float* white = (const float*)d_in[0];   // [2048, 40960]
    const float* black = (const float*)d_in[1];   // [2048, 40960]
    const float* stm   = (const float*)d_in[2];   // [2048, 1]
    const float* ftw   = (const float*)d_in[3];   // [256, 40960]
    const float* ftb   = (const float*)d_in[4];   // [256]
    const float* l1w   = (const float*)d_in[5];   // [32, 512]
    const float* l1b   = (const float*)d_in[6];   // [32]
    const float* l2w   = (const float*)d_in[7];   // [32, 32]
    const float* l2b   = (const float*)d_in[8];   // [32]
    const float* l3w   = (const float*)d_in[9];   // [1, 32]
    const float* l3b   = (const float*)d_in[10];  // [1]
    float* out = (float*)d_out;                   // [2048]

    // workspace layout (256-byte aligned blocks)
    char* ws = (char*)d_ws;
    _Float16* ftw_h  = (_Float16*)ws;                              // 20.97 MB
    _Float16* hidden = (_Float16*)(ws + (size_t)NN_H * NN_IN * 2); //  2.10 MB

    // 0) ft_w f32 -> f16 (stays hot in 192MB L2 for the GEMM)
    const int n8 = (NN_H * NN_IN) / 8;          // 1,310,720 vector-8 groups
    nnue_cvt_f16_kernel<<<(n8 + 255) / 256, 256, 0, stream>>>(ftw, ftw_h, n8);

    // 1) feature transformer GEMM (WMMA), both perspectives
    dim3 grid(NN_B / MROWS, 2);
    nnue_ft_wmma_kernel<<<grid, 512, 0, stream>>>(white, black, stm,
                                                  ftw_h, ftb, hidden);

    // 2) tail MLP
    nnue_tail_kernel<<<NN_B, 32, 0, stream>>>(hidden, l1w, l1b, l2w, l2b,
                                              l3w, l3b, out);
}
